// MambaFusionOperator_13580686590290
// MI455X (gfx1250) — matrix operand order repaired and checked by reading, not compiled, not run on hardware
//
#include <hip/hip_runtime.h>
#include <hip/hip_bf16.h>
#include <math.h>

typedef _Float16 v16h __attribute__((ext_vector_type(16)));
typedef _Float16 v8h  __attribute__((ext_vector_type(8)));
typedef float    v8f  __attribute__((ext_vector_type(8)));

#define TB    4            // B*T
#define HH    64
#define WW    64
#define NPIX  (HH*WW)
#define DMOD  48
#define DIN   96
#define DSTN  16
#define RANKN 3
#define NLAY  3
#define BINS  8
#define NSEQ  (TB*NPIX)    // 16384

#define MB_THREADS 128
#define SEQ_PER_WG 8
#define TOK        64      // SEQ_PER_WG * BINS
#define XDW        40      // padded xdbl row stride (160B, 16B aligned)

__device__ __forceinline__ float fast_rcp(float x)  { return __builtin_amdgcn_rcpf(x); }
__device__ __forceinline__ float siluf(float x)     { return x * fast_rcp(1.f + __expf(-x)); }
__device__ __forceinline__ float sigmf(float x)     { return fast_rcp(1.f + __expf(-x)); }
__device__ __forceinline__ float softplusf(float x) { return (x > 20.f) ? x : __logf(1.f + __expf(x)); }

// ---------------- WMMA fragment gathers (CDNA5 ISA 7.12.2 layouts) ----------------
// A 16x32 f16 (row-major in LDS, stride halves): lanes 0-15 hold M=0..15;
// half e: K = (e<8 ? e : e+8) + 8*(lane>=16)  ->  two contiguous 8-half runs.
__device__ __forceinline__ v16h frag_a(const _Float16* base, int stride, int mBase, int kBase, int lane) {
  const int m   = mBase + (lane & 15);
  const int kHi = (lane & 16) ? 8 : 0;
  const _Float16* p = base + m * stride + kBase + kHi;
  v8h lo = *(const v8h*)(p);        // K = kHi + 0..7
  v8h hi = *(const v8h*)(p + 16);   // K = kHi + 16..23
  return __builtin_shufflevector(lo, hi, 0,1,2,3,4,5,6,7,8,9,10,11,12,13,14,15);
}
// B 32x16 f16, weights stored N-major in LDS (s_w[n*kPad + k]):
// half e: K = (e<8 ? e : e+8) + 8*(lane>=16) -> two contiguous 8-half runs (matches frag_a).
__device__ __forceinline__ v16h frag_b(const _Float16* w, int kPad, int kBase, int nBase, int lane) {
  const int n   = nBase + (lane & 15);
  const int kHi = (lane & 16) ? 8 : 0;
  const _Float16* p = w + n * kPad + kBase + kHi;
  v8h lo = *(const v8h*)(p);
  v8h hi = *(const v8h*)(p + 16);
  return __builtin_shufflevector(lo, hi, 0,1,2,3,4,5,6,7,8,9,10,11,12,13,14,15);
}

// ---------------- front-end convs ----------------
__global__ __launch_bounds__(256) void k_rgb_conv1(const float* __restrict__ rgb,
                                                   const float* __restrict__ w,
                                                   const float* __restrict__ b,
                                                   float* __restrict__ out) {
  int gid = blockIdx.x * 256 + threadIdx.x;
  if (gid >= TB * DMOD * NPIX) return;
  int n = gid / (DMOD * NPIX);
  int r = gid % (DMOD * NPIX);
  int d = r / NPIX;
  int p = r % NPIX;
  int y = p / WW, x = p % WW;
  float acc = b[d];
  for (int c = 0; c < 3; ++c)
    for (int ky = 0; ky < 3; ++ky) {
      int yy = y + ky - 1; if (yy < 0 || yy >= HH) continue;
      for (int kx = 0; kx < 3; ++kx) {
        int xx = x + kx - 1; if (xx < 0 || xx >= WW) continue;
        acc += rgb[(n * 3 + c) * NPIX + yy * WW + xx] * w[((d * 3 + c) * 3 + ky) * 3 + kx];
      }
    }
  out[gid] = fmaxf(acc, 0.f);
}

__global__ __launch_bounds__(256) void k_conv48(const float* __restrict__ in,
                                                const float* __restrict__ w,
                                                const float* __restrict__ b,
                                                float* __restrict__ out, int relu) {
  int gid = blockIdx.x * 256 + threadIdx.x;
  if (gid >= TB * DMOD * NPIX) return;
  int n = gid / (DMOD * NPIX);
  int r = gid % (DMOD * NPIX);
  int d = r / NPIX;
  int p = r % NPIX;
  int y = p / WW, x = p % WW;
  float acc = b[d];
  for (int c = 0; c < DMOD; ++c) {
    const float* wp  = &w[((d * DMOD + c) * 3) * 3];
    const float* inp = &in[(n * DMOD + c) * NPIX];
    for (int ky = 0; ky < 3; ++ky) {
      int yy = y + ky - 1; if (yy < 0 || yy >= HH) continue;
      for (int kx = 0; kx < 3; ++kx) {
        int xx = x + kx - 1; if (xx < 0 || xx >= WW) continue;
        acc += inp[yy * WW + xx] * wp[ky * 3 + kx];
      }
    }
  }
  out[gid] = relu ? fmaxf(acc, 0.f) : acc;
}

// spike conv (1->48) + broadcast-add rgb context, write tokens [seq][bin][d]
__global__ __launch_bounds__(256) void k_spk_tokens(const float* __restrict__ spike,
                                                    const float* __restrict__ w,
                                                    const float* __restrict__ b,
                                                    const float* __restrict__ rgbctx,
                                                    float* __restrict__ tokens) {
  int gid = blockIdx.x * 256 + threadIdx.x;
  if (gid >= NSEQ * BINS * DMOD) return;
  int seq = gid / (BINS * DMOD);
  int r   = gid % (BINS * DMOD);
  int bin = r / DMOD, d = r % DMOD;
  int t = seq / NPIX, p = seq % NPIX;
  int y = p / WW, x = p % WW;
  float acc = b[d];
  for (int ky = 0; ky < 3; ++ky) {
    int yy = y + ky - 1; if (yy < 0 || yy >= HH) continue;
    for (int kx = 0; kx < 3; ++kx) {
      int xx = x + kx - 1; if (xx < 0 || xx >= WW) continue;
      acc += spike[(t * BINS + bin) * NPIX + yy * WW + xx] * w[d * 9 + ky * 3 + kx];
    }
  }
  acc += rgbctx[(t * DMOD + d) * NPIX + p];
  tokens[gid] = acc;
}

// ---------------- fused 3-layer Mamba stack (WMMA) ----------------
__global__ __launch_bounds__(MB_THREADS) void k_mamba(
    const float* __restrict__ tokens,
    const float* __restrict__ ln_g, const float* __restrict__ ln_b,
    const float* __restrict__ in_w, const float* __restrict__ conv_w,
    const float* __restrict__ conv_b, const float* __restrict__ xp_w,
    const float* __restrict__ dt_w, const float* __restrict__ dt_b,
    const float* __restrict__ A_log, const float* __restrict__ Dp,
    const float* __restrict__ out_w, float* __restrict__ pooled) {
  __shared__ float    s_seq[TOK][DMOD];     // residual stream (fp32)
  __shared__ _Float16 s_xh[TOK][64];        // LN output, K padded 48->64
  __shared__ _Float16 s_w[192 * 64];        // weight tile, N-major [n][k]
  __shared__ float    s_xin[TOK][DIN];      // xin, later reused as y
  __shared__ _Float16 s_zs[TOK][DIN];       // silu(z)
  __shared__ _Float16 s_xc[TOK][DIN];       // xc, later reused as y*silu(z)
  __shared__ float    s_xdbl[TOK][XDW];     // B(0..15) | C(16..31) | dt_raw(32..34)
  __shared__ float    s_A[DIN][DSTN];
  __shared__ float    s_dtw[DIN][RANKN];
  __shared__ float    s_dtb[DIN], s_Dp[DIN], s_cb[DIN];
  __shared__ float    s_cw[DIN][4];
  __shared__ float    s_lg[DMOD], s_lb[DMOD];

  const int tid  = threadIdx.x;
  const int lane = tid & 31;
  const int wave = tid >> 5;     // 4 waves -> 4 M-tiles of 16 tokens
  const int wg   = blockIdx.x;

  for (int i = tid; i < TOK * DMOD; i += MB_THREADS) {
    int tk = i / DMOD, d = i % DMOD;
    s_seq[tk][d] = tokens[(wg * TOK + tk) * DMOD + d];
  }

  for (int layer = 0; layer < NLAY; ++layer) {
    // ---- per-layer small params + in_w (f16, N-major, K padded to 64) ----
    for (int i = tid; i < DMOD; i += MB_THREADS) {
      s_lg[i] = ln_g[layer * DMOD + i];
      s_lb[i] = ln_b[layer * DMOD + i];
    }
    for (int i = tid; i < DIN; i += MB_THREADS) {
      s_dtb[i] = dt_b[layer * DIN + i];
      s_Dp[i]  = Dp[layer * DIN + i];
      s_cb[i]  = conv_b[layer * DIN + i];
    }
    for (int i = tid; i < DIN * DSTN; i += MB_THREADS)
      s_A[i / DSTN][i % DSTN] = -__expf(A_log[layer * DIN * DSTN + i]);
    for (int i = tid; i < DIN * RANKN; i += MB_THREADS)
      s_dtw[i / RANKN][i % RANKN] = dt_w[layer * DIN * RANKN + i];
    for (int i = tid; i < DIN * 4; i += MB_THREADS)
      s_cw[i / 4][i % 4] = conv_w[layer * DIN * 4 + i];
    for (int i = tid; i < 192 * 64; i += MB_THREADS) {   // [n][k] straight copy
      int n = i / 64, k = i % 64;
      s_w[i] = (k < DMOD) ? (_Float16)in_w[(layer * 192 + n) * DMOD + k] : (_Float16)0.f;
    }
    __syncthreads();

    // ---- layernorm -> f16 (one token per thread, waves 2/3 idle: wave-uniform) ----
    if (tid < TOK) {
      float m = 0.f;
      for (int k = 0; k < DMOD; ++k) m += s_seq[tid][k];
      m *= (1.f / DMOD);
      float v = 0.f;
      for (int k = 0; k < DMOD; ++k) { float dd = s_seq[tid][k] - m; v += dd * dd; }
      float rinv = __builtin_amdgcn_rsqf(v * (1.f / DMOD) + 1e-5f);
      for (int k = 0; k < DMOD; ++k)
        s_xh[tid][k] = (_Float16)(((s_seq[tid][k] - m) * rinv) * s_lg[k] + s_lb[k]);
      for (int k = DMOD; k < 64; ++k) s_xh[tid][k] = (_Float16)0.f;
    }
    __syncthreads();

    // ---- in-projection: (64 x 48pad64) @ (64 x 192), split into xin | silu(z) ----
    {
      const int mTile = wave;
      for (int nTile = 0; nTile < 12; ++nTile) {
        v8f acc = {};
#pragma unroll
        for (int ks = 0; ks < 2; ++ks) {
          v16h a  = frag_a(&s_xh[0][0], 64, mTile * 16, ks * 32, lane);
          v16h bb = frag_b(s_w, 64, ks * 32, nTile * 16, lane);
          acc = __builtin_amdgcn_wmma_f32_16x16x32_f16(false, a, false, bb, (short)0, acc, false, false);
        }
        const int n     = nTile * 16 + (lane & 15);
        const int mBase = mTile * 16 + ((lane & 16) ? 8 : 0);
        if (n < DIN) {                      // tile-aligned split: wave-uniform branch
#pragma unroll
          for (int r2 = 0; r2 < 8; ++r2) s_xin[mBase + r2][n] = acc[r2];
        } else {
#pragma unroll
          for (int r2 = 0; r2 < 8; ++r2) s_zs[mBase + r2][n - DIN] = (_Float16)siluf(acc[r2]);
        }
      }
    }
    __syncthreads();

    // ---- causal depthwise conv (DC=4) + silu -> xc (f16) ; load xp_w tile ----
    for (int i = tid; i < TOK * DIN; i += MB_THREADS) {
      int tk = i / DIN, c = i % DIN;
      int sl = tk >> 3, tp = tk & 7;
      float acc = s_cb[c];
#pragma unroll
      for (int j = 0; j < 4; ++j) {
        int q = tp - 3 + j;
        if (q >= 0) acc += s_cw[c][j] * s_xin[(sl << 3) + q][c];
      }
      s_xc[tk][c] = (_Float16)siluf(acc);
    }
    for (int i = tid; i < 48 * DIN; i += MB_THREADS) {   // N-major [n][k], K=96
      int n = i / DIN, k = i % DIN;
      s_w[i] = (n < 35) ? (_Float16)xp_w[(layer * 35 + n) * DIN + k] : (_Float16)0.f;
    }
    __syncthreads();

    // ---- xp-projection: (64 x 96) @ (96 x 48pad) -> B | C | dt_raw (permuted) ----
    {
      const int mTile = wave;
      for (int nTile = 0; nTile < 3; ++nTile) {
        v8f acc = {};
#pragma unroll
        for (int ks = 0; ks < 3; ++ks) {
          v16h a  = frag_a(&s_xc[0][0], DIN, mTile * 16, ks * 32, lane);
          v16h bb = frag_b(s_w, DIN, ks * 32, nTile * 16, lane);
          acc = __builtin_amdgcn_wmma_f32_16x16x32_f16(false, a, false, bb, (short)0, acc, false, false);
        }
        const int n     = nTile * 16 + (lane & 15);
        const int mBase = mTile * 16 + ((lane & 16) ? 8 : 0);
        if (n < 35) {
          const int col = (n < RANKN) ? (32 + n) : (n - RANKN);  // B:0-15, C:16-31, dt:32-34
#pragma unroll
          for (int r2 = 0; r2 < 8; ++r2) s_xdbl[mBase + r2][col] = acc[r2];
        }
      }
    }
    __syncthreads();

    // ---- selective scan: thread-private 16-wide state, 6 (seq,ch) pairs/thread ----
    for (int pp = 0; pp < 6; ++pp) {
      const int pair = tid + pp * MB_THREADS;      // 768 = 8 seq * 96 ch
      const int sl = pair / DIN, d = pair % DIN;
      float h[DSTN], Ac[DSTN];
#pragma unroll
      for (int s = 0; s < DSTN; ++s) { h[s] = 0.f; Ac[s] = s_A[d][s]; }
      const float w0 = s_dtw[d][0], w1 = s_dtw[d][1], w2 = s_dtw[d][2];
      const float bbs = s_dtb[d], dpar = s_Dp[d];
      for (int t = 0; t < BINS; ++t) {
        const int tk = (sl << 3) + t;
        const float4* row = (const float4*)&s_xdbl[tk][0];   // 16B-aligned (XDW=40)
        float Bv[DSTN], Cv[DSTN];
#pragma unroll
        for (int q = 0; q < 4; ++q) {
          ((float4*)Bv)[q] = row[q];
          ((float4*)Cv)[q] = row[4 + q];
        }
        float dt = softplusf(s_xdbl[tk][32] * w0 + s_xdbl[tk][33] * w1 + s_xdbl[tk][34] * w2 + bbs);
        float xv = (float)s_xc[tk][d];
        float dx = dt * xv;
        float yv = 0.f;
#pragma unroll
        for (int s = 0; s < DSTN; ++s) {
          h[s] = __expf(dt * Ac[s]) * h[s] + dx * Bv[s];
          yv  += h[s] * Cv[s];
        }
        s_xin[tk][d] = yv + xv * dpar;             // y (xin dead, reuse)
      }
    }
    __syncthreads();

    // ---- gate: gy = y * silu(z) -> f16 (reuse s_xc) ; load out_w tile ----
    for (int i = tid; i < TOK * DIN; i += MB_THREADS) {
      int tk = i / DIN, c = i % DIN;
      s_xc[tk][c] = (_Float16)(s_xin[tk][c] * (float)s_zs[tk][c]);
    }
    for (int i = tid; i < DMOD * DIN; i += MB_THREADS) {  // N-major [n][k], K=96
      int n = i / DIN, k = i % DIN;
      s_w[i] = (_Float16)out_w[(layer * DMOD + n) * DIN + k];
    }
    __syncthreads();

    // ---- out-projection: (64 x 96) @ (96 x 48), residual add into s_seq ----
    {
      const int mTile = wave;
      for (int nTile = 0; nTile < 3; ++nTile) {
        v8f acc = {};
#pragma unroll
        for (int ks = 0; ks < 3; ++ks) {
          v16h a  = frag_a(&s_xc[0][0], DIN, mTile * 16, ks * 32, lane);
          v16h bb = frag_b(s_w, DIN, ks * 32, nTile * 16, lane);
          acc = __builtin_amdgcn_wmma_f32_16x16x32_f16(false, a, false, bb, (short)0, acc, false, false);
        }
        const int n     = nTile * 16 + (lane & 15);
        const int mBase = mTile * 16 + ((lane & 16) ? 8 : 0);
#pragma unroll
        for (int r2 = 0; r2 < 8; ++r2) s_seq[mBase + r2][n] += acc[r2];
      }
    }
    __syncthreads();
  }

  // ---- mean over bins -> pooled NCHW (t, d, y, x) ----
  for (int i = tid; i < SEQ_PER_WG * DMOD; i += MB_THREADS) {
    int sl = i / DMOD, d = i % DMOD;
    float acc = 0.f;
#pragma unroll
    for (int t = 0; t < BINS; ++t) acc += s_seq[(sl << 3) + t][d];
    acc *= (1.f / BINS);
    int seq = wg * SEQ_PER_WG + sl;
    int timg = seq / NPIX, p = seq % NPIX;
    pooled[(timg * DMOD + d) * NPIX + p] = acc;
  }
}

// ---------------- head: 1x1 convs, sigmoid gate, blend ----------------
__global__ __launch_bounds__(256) void k_head(const float* __restrict__ rgb,
                                              const float* __restrict__ tmpc,
                                              const float* __restrict__ tmpg,
                                              const float* __restrict__ cw2, const float* __restrict__ cb2,
                                              const float* __restrict__ gw2, const float* __restrict__ gb2,
                                              float* __restrict__ out) {
  int gid = blockIdx.x * 256 + threadIdx.x;
  if (gid >= TB * NPIX) return;
  int t = gid / NPIX, p = gid % NPIX;
  float ac[3] = {cb2[0], cb2[1], cb2[2]};
  float ag[3] = {gb2[0], gb2[1], gb2[2]};
  for (int d = 0; d < DMOD; ++d) {
    float cv = tmpc[(t * DMOD + d) * NPIX + p];
    float gv = tmpg[(t * DMOD + d) * NPIX + p];
#pragma unroll
    for (int ch = 0; ch < 3; ++ch) {
      ac[ch] += cw2[ch * DMOD + d] * cv;
      ag[ch] += gw2[ch * DMOD + d] * gv;
    }
  }
#pragma unroll
  for (int ch = 0; ch < 3; ++ch) {
    int o = (t * 3 + ch) * NPIX + p;
    out[o] = rgb[o] + sigmf(ag[ch]) * ac[ch];
  }
}

extern "C" void kernel_launch(void* const* d_in, const int* in_sizes, int n_in,
                              void* d_out, int out_size, void* d_ws, size_t ws_size,
                              hipStream_t stream) {
  (void)in_sizes; (void)n_in; (void)out_size; (void)ws_size;
  const float* rgb     = (const float*)d_in[0];
  const float* spike   = (const float*)d_in[1];
  const float* rgb_w1  = (const float*)d_in[2];
  const float* rgb_b1  = (const float*)d_in[3];
  const float* rgb_w2  = (const float*)d_in[4];
  const float* rgb_b2  = (const float*)d_in[5];
  const float* spk_w   = (const float*)d_in[6];
  const float* spk_b   = (const float*)d_in[7];
  const float* ln_g    = (const float*)d_in[8];
  const float* ln_b    = (const float*)d_in[9];
  const float* in_w    = (const float*)d_in[10];
  const float* conv_w  = (const float*)d_in[11];
  const float* conv_b  = (const float*)d_in[12];
  const float* xp_w    = (const float*)d_in[13];
  const float* dt_w    = (const float*)d_in[14];
  const float* dt_b    = (const float*)d_in[15];
  const float* A_log   = (const float*)d_in[16];
  const float* Dp      = (const float*)d_in[17];
  const float* out_w   = (const float*)d_in[18];
  const float* cor_w1  = (const float*)d_in[19];
  const float* cor_b1  = (const float*)d_in[20];
  const float* cor_w2  = (const float*)d_in[21];
  const float* cor_b2  = (const float*)d_in[22];
  const float* gate_w1 = (const float*)d_in[23];
  const float* gate_b1 = (const float*)d_in[24];
  const float* gate_w2 = (const float*)d_in[25];
  const float* gate_b2 = (const float*)d_in[26];

  float* ws      = (float*)d_ws;
  float* tmp1    = ws;                              // 786432 f
  float* rgbctx  = ws + 786432;                     // 786432 f
  float* tokens  = ws + 2 * 786432;                 // 6291456 f
  float* pooled  = ws + 2 * 786432 + 6291456;       // 786432 f
  float* tmpc    = tmp1;                            // alias (dead after tokens)
  float* tmpg    = rgbctx;                          // alias (dead after tokens)

  k_rgb_conv1<<<3072, 256, 0, stream>>>(rgb, rgb_w1, rgb_b1, tmp1);
  k_conv48<<<3072, 256, 0, stream>>>(tmp1, rgb_w2, rgb_b2, rgbctx, 0);
  k_spk_tokens<<<(NSEQ * BINS * DMOD + 255) / 256, 256, 0, stream>>>(spike, spk_w, spk_b, rgbctx, tokens);
  k_mamba<<<NSEQ / SEQ_PER_WG, MB_THREADS, 0, stream>>>(tokens, ln_g, ln_b, in_w, conv_w, conv_b,
                                                        xp_w, dt_w, dt_b, A_log, Dp, out_w, pooled);
  k_conv48<<<3072, 256, 0, stream>>>(pooled, cor_w1, cor_b1, tmpc, 1);
  k_conv48<<<3072, 256, 0, stream>>>(pooled, gate_w1, gate_b1, tmpg, 1);
  k_head<<<(TB * NPIX + 255) / 256, 256, 0, stream>>>(rgb, tmpc, tmpg, cor_w2, cor_b2,
                                                      gate_w2, gate_b2, (float*)d_out);
}
